// DilatedAttention_29472065585688
// MI455X (gfx1250) — matrix operand order repaired and checked by reading, not compiled, not run on hardware
//
#include <hip/hip_runtime.h>
#include <math.h>
#include <stdint.h>

// ---------- types ----------
typedef __attribute__((ext_vector_type(16))) __bf16 bf16x16;
typedef __attribute__((ext_vector_type(8)))  __bf16 bf16x8;
typedef __attribute__((ext_vector_type(8)))  float  f32x8;

#define Bc 2
#define Lc 2048
#define Dc 1024
#define Hc 16
#define DHc 64
#define HIDc 4096
#define TOK (Bc*Lc)          // 4096 tokens
#define SUBL (Lc/2)          // 1024 per-parity subsequence length
#define WSUB 256             // window in subsequence units (512/2)

// ---------- helpers ----------
__device__ __forceinline__ float redmax16(float v) {
  v = fmaxf(v, __shfl_xor(v, 8, 32));
  v = fmaxf(v, __shfl_xor(v, 4, 32));
  v = fmaxf(v, __shfl_xor(v, 2, 32));
  v = fmaxf(v, __shfl_xor(v, 1, 32));
  return v;
}
__device__ __forceinline__ float redsum16(float v) {
  v += __shfl_xor(v, 8, 32);
  v += __shfl_xor(v, 4, 32);
  v += __shfl_xor(v, 2, 32);
  v += __shfl_xor(v, 1, 32);
  return v;
}
// Build a 16x32 bf16 A-fragment: lane m=lane&15, kh=lane>>4,
// element e -> K = kh*16 + e  => two contiguous 16B loads.
__device__ __forceinline__ bf16x16 load_afrag(const __bf16* rowptr, int kh) {
  bf16x8 lo = *(const bf16x8*)(rowptr + kh * 16);
  bf16x8 hi = *(const bf16x8*)(rowptr + kh * 16 + 8);
  bf16x16 a;
#pragma unroll
  for (int i = 0; i < 8; ++i) { a[i] = lo[i]; a[8 + i] = hi[i]; }
  return a;
}

// ---------- fp32 -> bf16 convert ----------
__global__ void cvt_bf16_kernel(const float* __restrict__ in, __bf16* __restrict__ out, int n) {
  for (int i = blockIdx.x * blockDim.x + threadIdx.x; i < n; i += gridDim.x * blockDim.x)
    out[i] = (__bf16)in[i];
}

// ---------- LayerNorm (fp32 in -> bf16 out) ----------
__global__ void ln_kernel(const float* __restrict__ x, const float* __restrict__ g,
                          const float* __restrict__ bta, __bf16* __restrict__ out) {
  const int row = blockIdx.x;
  const float* xr = x + (size_t)row * Dc;
  float s = 0.f, ss = 0.f;
  for (int c = threadIdx.x; c < Dc; c += blockDim.x) { float v = xr[c]; s += v; ss += v * v; }
  __shared__ float shs[256], shq[256];
  shs[threadIdx.x] = s; shq[threadIdx.x] = ss;
  __syncthreads();
  for (int off = 128; off > 0; off >>= 1) {
    if (threadIdx.x < off) { shs[threadIdx.x] += shs[threadIdx.x + off]; shq[threadIdx.x] += shq[threadIdx.x + off]; }
    __syncthreads();
  }
  const float mean = shs[0] * (1.f / Dc);
  const float var  = shq[0] * (1.f / Dc) - mean * mean;
  const float rs   = rsqrtf(var + 1e-5f);
  for (int c = threadIdx.x; c < Dc; c += blockDim.x)
    out[(size_t)row * Dc + c] = (__bf16)((xr[c] - mean) * rs * g[c] + bta[c]);
}

// ---------- WMMA GEMM: C[M,N] = A[M,K](bf16) * W[N,K]^T(bf16) + epilogue ----------
// Block = 4 waves covering a 64(M) x 64(N) tile. The shared 64x32 bf16 weight
// slab is DMA'd into LDS with GLOBAL_LOAD_ASYNC_TO_LDS_B128 (double buffered,
// ASYNCcnt-tracked) so the next slab's copy overlaps the current 16 WMMAs and
// B-side global traffic is shared across the 4 waves.
// MODE 0: qkv scatter into parity-split q/k and transposed v buffers (bf16)
// MODE 1: fp32 out = C + bias + resid   (O-proj and FF2 residual adds)
// MODE 2: bf16 out = gelu(C + bias)     (FF1)
template <int MODE>
__global__ void gemm16_kernel(const __bf16* __restrict__ A, const __bf16* __restrict__ W,
                              const float* __restrict__ bias, int N, int K, int Mtiles,
                              const float* __restrict__ resid, float* __restrict__ outf,
                              __bf16* __restrict__ outb, __bf16* __restrict__ qb,
                              __bf16* __restrict__ kb, __bf16* __restrict__ vt) {
  __shared__ __bf16 wl[2][64 * 32];                       // 2 x 4KB weight slabs
  const int tid  = threadIdx.x;                           // 0..127
  const int wib  = tid >> 5, lane = tid & 31;
  const int mb   = blockIdx.x % (Mtiles / 4);
  const int nt64 = blockIdx.x / (Mtiles / 4);
  const int m0 = (mb * 4 + wib) * 16, n0 = nt64 * 64;
  const int mrow = lane & 15, kh = lane >> 4;
  const unsigned ldsbase = (unsigned)(uintptr_t)(&wl[0][0]);  // addrspace(3) offset

  // Issue async DMA of W[n0..n0+63][k0..k0+31] into wl[buf] (256 x 16B granules,
  // 2 per thread). Tracked on ASYNCcnt; no VGPR data movement.
  auto issue_slab = [&](int k0, int buf) {
#pragma unroll
    for (int g = tid; g < 256; g += 128) {
      const int row = g >> 2;                    // 0..63
      const int c8  = (g & 3) * 8;               // bf16 column within slab
      const uint64_t src = (uint64_t)(uintptr_t)(W + (size_t)(n0 + row) * K + k0 + c8);
      const unsigned dst = ldsbase + (unsigned)(buf * 4096 + (row * 32 + c8) * 2);
      asm volatile("global_load_async_to_lds_b128 %0, %1, off"
                   :: "v"(dst), "v"(src) : "memory");
    }
  };

  f32x8 acc[4];
#pragma unroll
  for (int nt = 0; nt < 4; ++nt)
#pragma unroll
    for (int r = 0; r < 8; ++r) acc[nt][r] = 0.f;

  const __bf16* Arow = A + (size_t)(m0 + mrow) * K;
  issue_slab(0, 0);

  for (int k0 = 0; k0 < K; k0 += 32) {
    const int buf = (k0 >> 5) & 1;
    asm volatile("s_wait_asynccnt 0x0" ::: "memory");   // our slab granules landed
    __syncthreads();                                    // whole slab landed; prev reads done
    if (k0 + 32 < K) issue_slab(k0 + 32, buf ^ 1);      // overlap next DMA with WMMAs

    if (k0 + 32 < K) __builtin_prefetch(Arow + k0 + 32, 0, 1);   // global_prefetch_b8
    const bf16x16 af = load_afrag(Arow + k0, kh);
    const __bf16* wb = &wl[buf][0];
#pragma unroll
    for (int nt = 0; nt < 4; ++nt) {
      // B fragment (32x16) from LDS: lane holds column n=lane&15, K = kh*16 + e
      const bf16x16 bfr = *(const bf16x16*)(wb + (nt * 16 + mrow) * 32 + kh * 16);
      acc[nt] = __builtin_amdgcn_wmma_f32_16x16x32_bf16(false, af, false, bfr,
                                                        (short)0, acc[nt], false, false);
    }
  }

  // C layout: element r of v8f -> row M = r + 8*kh, col N = lane&15
#pragma unroll
  for (int nt = 0; nt < 4; ++nt) {
#pragma unroll
    for (int r = 0; r < 8; ++r) {
      const int gm = m0 + kh * 8 + r;
      const int gn = n0 + nt * 16 + mrow;
      float v = acc[nt][r] + bias[gn];
      if (MODE == 0) {
        const int b = gm >> 11, l = gm & (Lc - 1), p = l & 1, t = l >> 1;
        const int sel = gn >> 10, nn = gn & 1023, h = nn >> 6, dh = nn & 63;
        const size_t sub = (size_t)((b * Hc + h) * 2 + p);
        if (sel == 0)      qb[(sub * SUBL + t) * DHc + dh] = (__bf16)v;
        else if (sel == 1) kb[(sub * SUBL + t) * DHc + dh] = (__bf16)v;
        else               vt[(sub * DHc + dh) * SUBL + t] = (__bf16)v;  // transposed V
      } else if (MODE == 1) {
        outf[(size_t)gm * N + gn] = v + resid[(size_t)gm * N + gn];
      } else {
        const float ge = 0.5f * v * (1.0f + erff(v * 0.70710678118654752f));
        outb[(size_t)gm * N + gn] = (__bf16)ge;
      }
    }
  }
}

// ---------- Flash attention over dilated (parity-split) subsequences ----------
// One wave = one (b, h, parity, 16-query tile). Causal window WSUB=256 in
// subsequence coordinates == the reference's dilation-2 / window-512 mask.
__global__ void attn_kernel(const __bf16* __restrict__ qb, const __bf16* __restrict__ kb,
                            const __bf16* __restrict__ vt, __bf16* __restrict__ attn) {
  __shared__ __bf16 plds[4][16 * 32];
  const int wib = threadIdx.x >> 5, lane = threadIdx.x & 31;
  const int wid = blockIdx.x * 4 + wib;
  const int qt = wid & 63, p = (wid >> 6) & 1, h = (wid >> 7) & 15, b = wid >> 11;
  const size_t sub = (size_t)((b * Hc + h) * 2 + p);
  const __bf16* Q  = qb + sub * SUBL * DHc;
  const __bf16* Kb = kb + sub * SUBL * DHc;
  const __bf16* V  = vt + sub * DHc * SUBL;
  const int mrow = lane & 15, kh = lane >> 4;
  const int tq = qt * 16 + mrow;

  // Q fragments: 16x64 split into two 16x32 A fragments (loaded once)
  bf16x16 qf[2];
#pragma unroll
  for (int kk = 0; kk < 2; ++kk) qf[kk] = load_afrag(Q + tq * DHc + kk * 32, kh);

  // zero the K=16..31 half of the P staging tile once (P is only 16x16)
  __bf16* myp = plds[wib];
#pragma unroll
  for (int i = 0; i < 8; ++i) {
    const int e = lane * 8 + i;
    myp[(e >> 4) * 32 + 16 + (e & 15)] = (__bf16)0.f;
  }

  float mrun[8], lrun[8];
  f32x8 acc[4];
#pragma unroll
  for (int r = 0; r < 8; ++r) { mrun[r] = -1e30f; lrun[r] = 0.f; }
#pragma unroll
  for (int nt = 0; nt < 4; ++nt)
#pragma unroll
    for (int r = 0; r < 8; ++r) acc[nt][r] = 0.f;

  const int kt0 = (qt > 16) ? (qt - 16) : 0;
  for (int kt = kt0; kt <= qt; ++kt) {
    const int tk0 = kt * 16;
    // S = Q K^T : B fragment column n=lane&15 reads contiguous dh from K row
    f32x8 s;
#pragma unroll
    for (int r = 0; r < 8; ++r) s[r] = 0.f;
#pragma unroll
    for (int kk = 0; kk < 2; ++kk) {
      const bf16x16 bfr = *(const bf16x16*)(Kb + (size_t)(tk0 + mrow) * DHc + kk * 32 + kh * 16);
      s = __builtin_amdgcn_wmma_f32_16x16x32_bf16(false, qf[kk], false, bfr, (short)0, s, false, false);
    }
    // mask + scale + online softmax
    float pv[8], mnew[8], alpha[8];
#pragma unroll
    for (int r = 0; r < 8; ++r) {
      const int tqr = qt * 16 + kh * 8 + r;
      const int tkc = tk0 + mrow;
      const bool ok = (tkc <= tqr) && ((tqr - tkc) <= WSUB);
      pv[r] = ok ? s[r] * 0.125f : -1e30f;   // 1/sqrt(64)
    }
#pragma unroll
    for (int r = 0; r < 8; ++r) {
      const float mx = redmax16(pv[r]);
      mnew[r]  = fmaxf(mrun[r], mx);
      alpha[r] = expf(mrun[r] - mnew[r]);
      mrun[r]  = mnew[r];
      const float pe = expf(pv[r] - mnew[r]);
      lrun[r] = lrun[r] * alpha[r] + redsum16(pe);
      myp[(kh * 8 + r) * 32 + mrow] = (__bf16)pe;   // C-layout -> LDS
    }
#pragma unroll
    for (int nt = 0; nt < 4; ++nt)
#pragma unroll
      for (int r = 0; r < 8; ++r) acc[nt][r] *= alpha[r];

    asm volatile("s_wait_dscnt 0" ::: "memory");    // DS stores visible before re-read
    const bf16x16 pf = load_afrag(myp + mrow * 32, kh);  // A-layout P (K 16..31 are zero)
#pragma unroll
    for (int nt = 0; nt < 4; ++nt) {
      const bf16x16 vf = *(const bf16x16*)(V + (size_t)(nt * 16 + mrow) * SUBL + tk0 + kh * 16);
      acc[nt] = __builtin_amdgcn_wmma_f32_16x16x32_bf16(false, pf, false, vf, (short)0, acc[nt], false, false);
    }
  }

  // normalize and scatter back to token order as bf16 A-matrix for O-proj
#pragma unroll
  for (int nt = 0; nt < 4; ++nt)
#pragma unroll
    for (int r = 0; r < 8; ++r) {
      const int trow = qt * 16 + kh * 8 + r;
      const float o = acc[nt][r] / lrun[r];
      const size_t gm = (size_t)b * Lc + (2 * trow + p);
      const int gn = h * DHc + nt * 16 + mrow;
      attn[gm * Dc + gn] = (__bf16)o;
    }
}

// ---------- host launcher ----------
extern "C" void kernel_launch(void* const* d_in, const int* in_sizes, int n_in,
                              void* d_out, int out_size, void* d_ws, size_t ws_size,
                              hipStream_t stream) {
  const float* x      = (const float*)d_in[0];
  const float* ln1_g  = (const float*)d_in[1];
  const float* ln1_b  = (const float*)d_in[2];
  const float* w_qkv  = (const float*)d_in[3];
  const float* b_qkv  = (const float*)d_in[4];
  const float* w_o    = (const float*)d_in[5];
  const float* b_o    = (const float*)d_in[6];
  const float* ln2_g  = (const float*)d_in[7];
  const float* ln2_b  = (const float*)d_in[8];
  const float* w_ff1  = (const float*)d_in[9];
  const float* b_ff1  = (const float*)d_in[10];
  const float* w_ff2  = (const float*)d_in[11];
  const float* b_ff2  = (const float*)d_in[12];

  char* wsp = (char*)d_ws;
  auto carve = [&](size_t bytes) -> char* {
    char* r = wsp;
    wsp += (bytes + 255) & ~(size_t)255;
    return r;
  };
  __bf16* wqkv_b = (__bf16*)carve((size_t)3 * Dc * Dc * 2);
  __bf16* wo_b   = (__bf16*)carve((size_t)Dc * Dc * 2);
  __bf16* wff1_b = (__bf16*)carve((size_t)HIDc * Dc * 2);
  __bf16* wff2_b = (__bf16*)carve((size_t)Dc * HIDc * 2);
  __bf16* h1_b   = (__bf16*)carve((size_t)TOK * Dc * 2);
  __bf16* q_b    = (__bf16*)carve((size_t)TOK * Dc * 2);
  __bf16* k_b    = (__bf16*)carve((size_t)TOK * Dc * 2);
  __bf16* v_t    = (__bf16*)carve((size_t)TOK * Dc * 2);
  __bf16* attn_b = (__bf16*)carve((size_t)TOK * Dc * 2);
  float*  out1_f = (float*)carve((size_t)TOK * Dc * 4);
  __bf16* h2_b   = (__bf16*)carve((size_t)TOK * Dc * 2);
  __bf16* ff_b   = (__bf16*)carve((size_t)TOK * HIDc * 2);

  // weight conversions (deterministic each call)
  cvt_bf16_kernel<<<2048, 256, 0, stream>>>(w_qkv, wqkv_b, 3 * Dc * Dc);
  cvt_bf16_kernel<<<2048, 256, 0, stream>>>(w_o,   wo_b,   Dc * Dc);
  cvt_bf16_kernel<<<2048, 256, 0, stream>>>(w_ff1, wff1_b, HIDc * Dc);
  cvt_bf16_kernel<<<2048, 256, 0, stream>>>(w_ff2, wff2_b, Dc * HIDc);

  // LN1
  ln_kernel<<<TOK, 256, 0, stream>>>(x, ln1_g, ln1_b, h1_b);

  // QKV: M=4096 N=3072 K=1024 -> 64 m-blocks * 48 n-blocks
  gemm16_kernel<0><<<(TOK / 64) * (3 * Dc / 64), 128, 0, stream>>>(
      h1_b, wqkv_b, b_qkv, 3 * Dc, Dc, TOK / 16, nullptr, nullptr, nullptr, q_b, k_b, v_t);

  // flash attention: 4096 waves / 4
  attn_kernel<<<1024, 128, 0, stream>>>(q_b, k_b, v_t, attn_b);

  // O-proj + residual(x) -> out1 (fp32)
  gemm16_kernel<1><<<(TOK / 64) * (Dc / 64), 128, 0, stream>>>(
      attn_b, wo_b, b_o, Dc, Dc, TOK / 16, x, out1_f, nullptr, nullptr, nullptr, nullptr);

  // LN2
  ln_kernel<<<TOK, 256, 0, stream>>>(out1_f, ln2_g, ln2_b, h2_b);

  // FF1 + GELU -> bf16
  gemm16_kernel<2><<<(TOK / 64) * (HIDc / 64), 128, 0, stream>>>(
      h2_b, wff1_b, b_ff1, HIDc, Dc, TOK / 16, nullptr, nullptr, ff_b, nullptr, nullptr, nullptr);

  // FF2 + residual(out1) -> d_out (fp32)
  gemm16_kernel<1><<<(TOK / 64) * (Dc / 64), 128, 0, stream>>>(
      ff_b, wff2_b, b_ff2, Dc, HIDc, TOK / 16, out1_f, (float*)d_out, nullptr, nullptr, nullptr, nullptr);

  (void)in_sizes; (void)n_in; (void)out_size; (void)ws_size;
}
